// UFourierLayer_34918084116740
// MI455X (gfx1250) — compile-verified, matrix operand order checked by reading, not run on hardware
//
#include <hip/hip_runtime.h>
#include <hip/hip_bf16.h>

// ---------------------------------------------------------------------------
// Sizes (match reference)
// ---------------------------------------------------------------------------
#define B_DIM  32
#define C_DIM  128
#define L_DIM  4096
#define T_DIM  512
#define KTOP   8
#define NFREQ  (L_DIM / 2 + 1)   // 2049
#define NTH    256               // threads per FFT block (8 waves)

typedef __attribute__((ext_vector_type(16))) _Float16 v16h;
typedef __attribute__((ext_vector_type(8)))  float    v8f;

// ---------------------------------------------------------------------------
// CDNA5 async global -> LDS copy (ASYNCcnt-tracked, no VGPR staging).
// GVS mode: mem = SADDR(64b) + VADDR(u32 offset); VDST = LDS byte address.
// ---------------------------------------------------------------------------
__device__ __forceinline__ void async_b128_to_lds(unsigned lds_addr,
                                                  unsigned long long gbase,
                                                  unsigned goff) {
    asm volatile("global_load_async_to_lds_b128 %0, %1, %2"
                 :
                 : "v"(lds_addr), "v"(goff), "s"(gbase)
                 : "memory");
}
__device__ __forceinline__ void wait_asynccnt0() {
    asm volatile("s_wait_asynccnt 0x0" ::: "memory");
}

// ---------------------------------------------------------------------------
// Kernel A: scale[b][c] = 1 + tanh(time_emb @ W^T + bias)   via WMMA f16
//   One wave computes one 16x16 tile; K=512 fully unrolled -> 16 WMMAs.
// ---------------------------------------------------------------------------
__global__ __launch_bounds__(32) void scale_wmma_kernel(
    const float* __restrict__ time_emb,   // (32, 512)
    const float* __restrict__ W,          // (128, 512)
    const float* __restrict__ bias,       // (128)
    float* __restrict__ scale_out)        // (32, 128)
{
    const int tile_m = blockIdx.x >> 3;   // 0..1  (batch tiles)
    const int tile_n = blockIdx.x & 7;    // 0..7  (channel tiles)
    const int lane   = threadIdx.x;       // 0..31 (wave32)
    const int half   = lane >> 4;         // 0 | 1
    const int l16    = lane & 15;

    const int mrow = tile_m * 16 + l16;   // batch row for A
    const int ncol = tile_n * 16 + l16;   // channel col for B

    v8f acc = {};
#pragma unroll
    for (int k0 = 0; k0 < T_DIM; k0 += 32) {
        v16h a, bm;
        // ISA 16-bit A layout (16x32):
        //   element j of lane: K = (j/8)*16 + (lane/16)*8 + (j%8), M = lane%16
        // B (32x16) mirrors with N = lane%16 and the same K mapping.
#pragma unroll
        for (int j = 0; j < 16; ++j) {
            const int kk = k0 + ((j >> 3) << 4) + (half << 3) + (j & 7);
            a[j]  = (_Float16)time_emb[mrow * T_DIM + kk];
            bm[j] = (_Float16)W[ncol * T_DIM + kk];   // B[k][n] = W[n][k]
        }
        // 8 args: (neg_a, A, neg_b, B, c_mod, C, reuse_a, reuse_b)
        acc = __builtin_amdgcn_wmma_f32_16x16x32_f16(
            false, a, false, bm, (short)0, acc, false, false);
    }

    // C/D layout: element v of lane: M = v + (lane/16)*8, N = lane%16
#pragma unroll
    for (int v = 0; v < 8; ++v) {
        const int m = tile_m * 16 + v + (half << 3);
        const int n = tile_n * 16 + l16;
        const float val = acc[v] + bias[n];
        scale_out[m * C_DIM + n] = 1.0f + tanhf(val);
    }
}

// ---------------------------------------------------------------------------
// Kernel B: per-row (b,c):
//   1) async-copy x row (16 KB) global->LDS, overlapped with twiddle build
//   2) in-place bit-reversal fused with modulation scale
//   3) radix-2 DIT FFT in LDS (12 stages), table twiddles
//   4) squared magnitudes, iterative argmax -> top-8
//   5) direct sparse inverse (8 bins), vectorized B128 stores
// ---------------------------------------------------------------------------
__global__ __launch_bounds__(NTH) void fft_topk_kernel(
    const float* __restrict__ x,      // (B*C, L)
    const float* __restrict__ scale,  // (B*C)
    float* __restrict__ out)          // (B*C, L)
{
    __shared__ float re[L_DIM];
    __shared__ float im[L_DIM];
    __shared__ float twc[L_DIM / 2];   // cos(2*pi*t/N)
    __shared__ float tws[L_DIM / 2];   // sin(2*pi*t/N)
    __shared__ float amp[NFREQ];
    __shared__ float redV[NTH];
    __shared__ int   redI[NTH];
    __shared__ int   topIdx[KTOP];
    __shared__ float topRe[KTOP];
    __shared__ float topIm[KTOP];

    const int row = blockIdx.x;            // b * C + c
    const int tid = threadIdx.x;
    const float s = scale[row];
    const float* __restrict__ xr = x + (size_t)row * L_DIM;

    // ---- 1) kick off async row copy into re[] (linear), 4 x B128 per thread
    {
        const unsigned lds_re = (unsigned)(uintptr_t)(&re[0]);
        const unsigned long long gb = (unsigned long long)(uintptr_t)xr;
#pragma unroll
        for (int it = 0; it < 4; ++it) {
            const unsigned off = ((unsigned)tid << 4) + ((unsigned)it << 12);
            async_b128_to_lds(lds_re + off, gb, off);
        }
    }

    // ---- overlap: twiddle table (2048 sincos per block)
    const float twoPiN = 6.28318530717958647692f / (float)L_DIM;
    for (int t = tid; t < L_DIM / 2; t += NTH) {
        float sv, cv;
        __sincosf(twoPiN * (float)t, &sv, &cv);
        twc[t] = cv;
        tws[t] = sv;
    }

    wait_asynccnt0();
    __syncthreads();

    // ---- 2) in-place bit-reversal (12-bit) fused with modulation; zero im
    for (int i = tid; i < L_DIM; i += NTH) {
        im[i] = 0.0f;
        const int r = (int)(__brev((unsigned)i) >> 20);
        if (r > i) {               // pair {i,r} owned by this thread only
            const float a = re[i], b = re[r];
            re[i] = b * s;
            re[r] = a * s;
        } else if (r == i) {
            re[i] *= s;
        }
    }
    __syncthreads();

    // ---- 3) radix-2 DIT, twiddle = e^{-i*2pi*j/len} = (twc[j*ts], -tws[j*ts])
    for (int len = 2; len <= L_DIM; len <<= 1) {
        const int half  = len >> 1;
        const int tstep = L_DIM / len;
        for (int t = tid; t < (L_DIM >> 1); t += NTH) {
            const int j  = t & (half - 1);
            const int i0 = ((t & ~(half - 1)) << 1) | j;
            const int i1 = i0 + half;
            const float c  = twc[j * tstep];
            const float sn = tws[j * tstep];
            const float x1r = re[i1], x1i = im[i1];
            const float vr = x1r * c + x1i * sn;     // x1 * e^{-i theta}
            const float vi = x1i * c - x1r * sn;
            const float ur = re[i0], ui = im[i0];
            re[i0] = ur + vr;  im[i0] = ui + vi;
            re[i1] = ur - vr;  im[i1] = ui - vi;
        }
        __syncthreads();
    }

    // ---- 4) squared magnitude (order-equivalent to abs for top-k)
    for (int f = tid; f < NFREQ; f += NTH)
        amp[f] = re[f] * re[f] + im[f] * im[f];
    __syncthreads();

    // iterative argmax for top-8
    for (int kk = 0; kk < KTOP; ++kk) {
        float bestV = -1.0f;
        int   bestI = 0;
        for (int f = tid; f < NFREQ; f += NTH) {
            const float v = amp[f];
            if (v > bestV) { bestV = v; bestI = f; }
        }
        redV[tid] = bestV;
        redI[tid] = bestI;
        __syncthreads();
        for (int sft = NTH >> 1; sft > 0; sft >>= 1) {
            if (tid < sft) {
                if (redV[tid + sft] > redV[tid]) {
                    redV[tid] = redV[tid + sft];
                    redI[tid] = redI[tid + sft];
                }
            }
            __syncthreads();
        }
        if (tid == 0) {
            const int fi = redI[0];
            topIdx[kk] = fi;
            topRe[kk]  = re[fi];
            topIm[kk]  = im[fi];
            amp[fi]    = -2.0f;   // mask out
        }
        __syncthreads();
    }

    // ---- 5) direct sparse inverse rFFT; each thread -> 16 consecutive n,
    //         phase recurrence m += f (mod N), four B128 stores.
    float* __restrict__ orow = out + (size_t)row * L_DIM;
    const float invN = 1.0f / (float)L_DIM;
    {
        const int n0 = tid << 4;
        float accv[16];
#pragma unroll
        for (int j = 0; j < 16; ++j) accv[j] = 0.0f;

#pragma unroll
        for (int kk = 0; kk < KTOP; ++kk) {
            const int f = topIdx[kk];
            const float w  = (f == 0 || f == (L_DIM / 2)) ? 1.0f : 2.0f;
            const float wr = w * topRe[kk];
            const float wi = w * topIm[kk];
            int m = (f * n0) & (L_DIM - 1);
#pragma unroll
            for (int j = 0; j < 16; ++j) {
                const int mm = m & (L_DIM / 2 - 1);
                const float sgn = (m & (L_DIM / 2)) ? -1.0f : 1.0f;  // +pi shift
                accv[j] += sgn * (wr * twc[mm] - wi * tws[mm]);
                m = (m + f) & (L_DIM - 1);
            }
        }
#pragma unroll
        for (int j = 0; j < 16; j += 4) {
            float4 v = make_float4(accv[j]     * invN, accv[j + 1] * invN,
                                   accv[j + 2] * invN, accv[j + 3] * invN);
            *(float4*)(&orow[n0 + j]) = v;
        }
    }
}

// ---------------------------------------------------------------------------
// Launch
//   d_in order: x (B*C*L), time_emb (B*T), W (C*T), b (C)
//   d_out: (B, C, L) f32;  d_ws: scale buffer (B*C f32 = 16 KB)
// ---------------------------------------------------------------------------
extern "C" void kernel_launch(void* const* d_in, const int* in_sizes, int n_in,
                              void* d_out, int out_size, void* d_ws, size_t ws_size,
                              hipStream_t stream) {
    const float* x        = (const float*)d_in[0];
    const float* time_emb = (const float*)d_in[1];
    const float* W        = (const float*)d_in[2];
    const float* bias     = (const float*)d_in[3];
    float* out            = (float*)d_out;
    float* scale_ws       = (float*)d_ws;

    scale_wmma_kernel<<<dim3(16), dim3(32), 0, stream>>>(time_emb, W, bias, scale_ws);
    fft_topk_kernel<<<dim3(B_DIM * C_DIM), dim3(NTH), 0, stream>>>(x, scale_ws, out);
}
